// ItemTransformerEncoder_54443005444570
// MI455X (gfx1250) — matrix operand order repaired and checked by reading, not compiled, not run on hardware
//
#include <hip/hip_runtime.h>
#include <math.h>

// ---------------------------------------------------------------------------
// Types for CDNA5 WMMA (wave32)
// ---------------------------------------------------------------------------
typedef __attribute__((ext_vector_type(16))) _Float16 v16h;
typedef __attribute__((ext_vector_type(8)))  _Float16 v8h;
typedef __attribute__((ext_vector_type(8)))  float    v8f;

#define N_NODES 50000
#define DIM_C   128
#define HEADS_C 2
#define DMLP_C  512
#define E_EDGES 800000
#define DEPTH_C 2
#define NEG_SLOPE 0.2f

// ---------------------------------------------------------------------------
// Helpers
// ---------------------------------------------------------------------------
__device__ __forceinline__ float wave_sum(float v) {
#pragma unroll
  for (int off = 16; off > 0; off >>= 1) v += __shfl_xor(v, off, 32);
  return v;
}

// order-preserving float <-> uint key for atomicMax on signed floats
__device__ __forceinline__ unsigned fkey(float f) {
  unsigned u = __float_as_uint(f);
  return (u & 0x80000000u) ? ~u : (u | 0x80000000u);
}
__device__ __forceinline__ float funkey(unsigned k) {
  unsigned u = (k & 0x80000000u) ? (k & 0x7FFFFFFFu) : ~k;
  return __uint_as_float(u);
}

// ---------------------------------------------------------------------------
// Pack f32 weights W[K x Nout] into per-lane WMMA B-fragments (f16):
//   Wpk[ ((tn*(K/32) + ks)*32 + lane)*16 + j ]
//     = W[ (ks*32 + (lane>=16 ? 8:0) + j + (j&8)) * Nout + tn*16 + (lane&15) ]
// so the GEMM loads one contiguous 32B v16h per lane per (tile, k-step).
// ---------------------------------------------------------------------------
__global__ void pack_w(const float* __restrict__ W, _Float16* __restrict__ Wpk,
                       int K, int Nout) {
  int i = blockIdx.x * 256 + threadIdx.x;
  if (i >= K * Nout) return;
  int j    = i & 15;
  int lane = (i >> 4) & 31;
  int rest = i >> 9;
  int ksteps = K >> 5;
  int ks = rest % ksteps;
  int tn = rest / ksteps;
  int bK = (lane >> 4) ? 8 : 0;
  int n  = tn * 16 + (lane & 15);
  int k  = ks * 32 + bK + j + (j & 8);
  Wpk[i] = (_Float16)W[(size_t)k * Nout + n];
}

// ---------------------------------------------------------------------------
// LayerNorm over rows of 128, emit f16 for WMMA A-operand. One wave per row.
// ---------------------------------------------------------------------------
__global__ void layernorm_f16(const float* __restrict__ x,
                              const float* __restrict__ g,
                              const float* __restrict__ b,
                              _Float16* __restrict__ out, int nrows) {
  int row  = blockIdx.x * 8 + (threadIdx.x >> 5);
  int lane = threadIdx.x & 31;
  if (row >= nrows) return;
  const float* xr = x + (size_t)row * DIM_C;
  float v[4];
  float s = 0.f;
#pragma unroll
  for (int i = 0; i < 4; ++i) { v[i] = xr[lane + 32 * i]; s += v[i]; }
  s = wave_sum(s);
  float mu = s * (1.0f / DIM_C);
  float var = 0.f;
#pragma unroll
  for (int i = 0; i < 4; ++i) { float d = v[i] - mu; var += d * d; }
  var = wave_sum(var) * (1.0f / DIM_C);
  float rs = rsqrtf(var + 1e-5f);
  _Float16* orow = out + (size_t)row * DIM_C;
#pragma unroll
  for (int i = 0; i < 4; ++i) {
    int c = lane + 32 * i;
    orow[c] = (_Float16)((v[i] - mu) * rs * g[c] + b[c]);
  }
}

// ---------------------------------------------------------------------------
// WMMA GEMM, K templated (fully unrolled), 16x32 output per wave (A reused
// across two B tiles -> two v_wmma per K-step).
//   mode 0: store f32           mode 1: residual add into f32 (x += ...)
//   mode 2: exact GELU -> f16
// ---------------------------------------------------------------------------
template <int K, int MODE>
__global__ void gemm_wmma_f16(const _Float16* __restrict__ A,
                              const _Float16* __restrict__ Wpk,
                              const float* __restrict__ bias,
                              float* __restrict__ outF,
                              _Float16* __restrict__ outH,
                              int M, int Nout) {
  constexpr int KS = K >> 5;                 // k-steps of 32
  const int lane = threadIdx.x & 31;
  const int wave = threadIdx.x >> 5;
  const int tiles_n2 = Nout >> 5;            // pairs of 16-wide tiles
  const int pair = blockIdx.x * 8 + wave;
  if (pair >= (M >> 4) * tiles_n2) return;
  const int tm = pair / tiles_n2, tn2 = pair % tiles_n2;
  const int row0 = tm << 4, n0 = tn2 << 5;

  // A layout (16-bit 16x32): lanes 0-15 -> K 0-7 & 16-23; lanes 16-31 -> K 8-15 & 24-31
  const size_t aBase = (size_t)(row0 + (lane & 15)) * K + ((lane >> 4) ? 8 : 0);
  // packed B fragment bases for the two 16-wide tiles
  const _Float16* w0 = Wpk + ((size_t)(2 * tn2) * KS * 32 + lane) * 16;
  const _Float16* w1 = Wpk + ((size_t)(2 * tn2 + 1) * KS * 32 + lane) * 16;

  v8f c0 = {}, c1 = {};
#pragma unroll
  for (int ks = 0; ks < KS; ++ks) {
    v8h alo = *(const v8h*)(A + aBase + ks * 32);
    v8h ahi = *(const v8h*)(A + aBase + ks * 32 + 16);
    v16h a = __builtin_shufflevector(alo, ahi, 0, 1, 2, 3, 4, 5, 6, 7,
                                     8, 9, 10, 11, 12, 13, 14, 15);
    v16h b0 = *(const v16h*)(w0 + (size_t)ks * 32 * 16);
    v16h b1 = *(const v16h*)(w1 + (size_t)ks * 32 * 16);
    c0 = __builtin_amdgcn_wmma_f32_16x16x32_f16(false, a, false, b0,
                                                (short)0, c0, false, false);
    c1 = __builtin_amdgcn_wmma_f32_16x16x32_f16(false, a, false, b1,
                                                (short)0, c1, false, false);
  }

  // C/D layout: lane n<16 -> col n, rows row0..+7; lane>=16 -> rows row0+8..+15
  const int r0 = row0 + ((lane >> 4) ? 8 : 0);
#pragma unroll
  for (int t = 0; t < 2; ++t) {
    const int cn = n0 + t * 16 + (lane & 15);
    const float bb = bias[cn];
    const v8f& c = t ? c1 : c0;
    if (MODE == 0) {
#pragma unroll
      for (int i = 0; i < 8; ++i)
        outF[(size_t)(r0 + i) * Nout + cn] = c[i] + bb;
    } else if (MODE == 1) {
#pragma unroll
      for (int i = 0; i < 8; ++i) {
        size_t idx = (size_t)(r0 + i) * Nout + cn;
        outF[idx] += c[i] + bb;
      }
    } else {
#pragma unroll
      for (int i = 0; i < 8; ++i) {
        float v = c[i] + bb;
        float gl = 0.5f * v * (1.0f + erff(v * 0.70710678118654752f));
        outH[(size_t)(r0 + i) * Nout + cn] = (_Float16)gl;
      }
    }
  }
}

// ---------------------------------------------------------------------------
// Edge pass 1: scores s[e,h] = sum_c lrelu(xl[src]+xr[dst]) * att[h,c];
// atomicMax into per-(dst,head) running max. One wave per edge.
// ---------------------------------------------------------------------------
__global__ void edge_score(const int* __restrict__ srcI, const int* __restrict__ dstI,
                           const float* __restrict__ xl, const float* __restrict__ xr,
                           const float* __restrict__ att,
                           float* __restrict__ sbuf, unsigned* __restrict__ smax) {
  int e = blockIdx.x * 8 + (threadIdx.x >> 5);
  int lane = threadIdx.x & 31;
  if (e >= E_EDGES) return;
  int s = srcI[e], d = dstI[e];
  const float* pl = xl + (size_t)s * (HEADS_C * DIM_C);
  const float* pr = xr + (size_t)d * (HEADS_C * DIM_C);
  __builtin_prefetch(pl, 0, 1);
  __builtin_prefetch(pr, 0, 1);
#pragma unroll
  for (int h = 0; h < HEADS_C; ++h) {
    const float* ph = att + h * DIM_C;
    float acc = 0.f;
#pragma unroll
    for (int i = 0; i < 4; ++i) {
      int c = lane + 32 * i;
      float m = pl[h * DIM_C + c] + pr[h * DIM_C + c];
      m = (m > 0.f) ? m : NEG_SLOPE * m;
      acc += m * ph[c];
    }
    acc = wave_sum(acc);
    if (lane == 0) {
      sbuf[(size_t)e * HEADS_C + h] = acc;
      atomicMax(&smax[(size_t)d * HEADS_C + h], fkey(acc));
    }
  }
}

// ---------------------------------------------------------------------------
// Edge pass 2: ex = exp(s - smax[dst]); overwrite sbuf; atomicAdd denom.
// One thread per (edge, head).
// ---------------------------------------------------------------------------
__global__ void edge_exp(const int* __restrict__ dstI,
                         float* __restrict__ sbuf,
                         const unsigned* __restrict__ smax,
                         float* __restrict__ denom) {
  int i = blockIdx.x * 256 + threadIdx.x;
  if (i >= E_EDGES * HEADS_C) return;
  int e = i >> 1, h = i & 1;
  int d = dstI[e];
  float mx = funkey(smax[(size_t)d * HEADS_C + h]);
  float ex = expf(sbuf[i] - mx);
  sbuf[i] = ex;
  atomicAdd(&denom[(size_t)d * HEADS_C + h], ex);
}

// ---------------------------------------------------------------------------
// Edge pass 3: agg[dst] += (ex/denom) * xl[src]. One wave per edge, 8 ch/lane.
// ---------------------------------------------------------------------------
__global__ void edge_agg(const int* __restrict__ srcI, const int* __restrict__ dstI,
                         const float* __restrict__ xl,
                         const float* __restrict__ sbuf,
                         const float* __restrict__ denom,
                         float* __restrict__ agg) {
  int e = blockIdx.x * 8 + (threadIdx.x >> 5);
  int lane = threadIdx.x & 31;
  if (e >= E_EDGES) return;
  int s = srcI[e], d = dstI[e];
  float a0 = sbuf[(size_t)e * 2 + 0] / (denom[(size_t)d * 2 + 0] + 1e-16f);
  float a1 = sbuf[(size_t)e * 2 + 1] / (denom[(size_t)d * 2 + 1] + 1e-16f);
  const float* pl = xl + (size_t)s * (HEADS_C * DIM_C);
  float* pd = agg + (size_t)d * (HEADS_C * DIM_C);
#pragma unroll
  for (int i = 0; i < 8; ++i) {
    int ch = lane + 32 * i;                 // 0..255 ; head = ch/128
    float al = (ch < DIM_C) ? a0 : a1;
    atomicAdd(&pd[ch], al * pl[ch]);
  }
}

// ---------------------------------------------------------------------------
// gat16 = f16(agg + gat_b)  (prep A-operand for Wp GEMM)
// ---------------------------------------------------------------------------
__global__ void gat_finalize(const float* __restrict__ agg,
                             const float* __restrict__ gbias,
                             _Float16* __restrict__ out, int n) {
  int i = blockIdx.x * 256 + threadIdx.x;
  if (i < n) out[i] = (_Float16)(agg[i] + gbias[i & (HEADS_C * DIM_C - 1)]);
}

// ---------------------------------------------------------------------------
// Host orchestration
// ---------------------------------------------------------------------------
static inline size_t al256(size_t x) { return (x + 255) & ~(size_t)255; }

extern "C" void kernel_launch(void* const* d_in, const int* in_sizes, int n_in,
                              void* d_out, int out_size, void* d_ws, size_t ws_size,
                              hipStream_t stream) {
  (void)in_sizes; (void)n_in; (void)out_size; (void)ws_size;
  const float* x_in  = (const float*)d_in[0];
  const int*   eidx  = (const int*)d_in[1];
  const float* ln1_g = (const float*)d_in[2];
  const float* ln1_b = (const float*)d_in[3];
  const float* Wl    = (const float*)d_in[4];
  const float* bl    = (const float*)d_in[5];
  const float* Wr    = (const float*)d_in[6];
  const float* br    = (const float*)d_in[7];
  const float* att   = (const float*)d_in[8];
  const float* gat_b = (const float*)d_in[9];
  const float* Wp    = (const float*)d_in[10];
  const float* bp    = (const float*)d_in[11];
  const float* ln2_g = (const float*)d_in[12];
  const float* ln2_b = (const float*)d_in[13];
  const float* W1    = (const float*)d_in[14];
  const float* b1    = (const float*)d_in[15];
  const float* W2    = (const float*)d_in[16];
  const float* b2    = (const float*)d_in[17];

  const int* srcI = eidx;            // edge_index[0]
  const int* dstI = eidx + E_EDGES;  // edge_index[1]

  // workspace carve-up
  char* ws = (char*)d_ws;
  size_t off = 0;
  auto carve = [&](size_t bytes) -> void* { void* p = ws + off; off = al256(off + bytes); return p; };
  float*     xbuf  = (float*)    carve((size_t)N_NODES * DIM_C * 4);
  _Float16*  h16   = (_Float16*) carve((size_t)N_NODES * DIM_C * 2);
  float*     xl    = (float*)    carve((size_t)N_NODES * HEADS_C * DIM_C * 4);
  float*     xr    = (float*)    carve((size_t)N_NODES * HEADS_C * DIM_C * 4);
  float*     sbuf  = (float*)    carve((size_t)E_EDGES * HEADS_C * 4);
  unsigned*  smax  = (unsigned*) carve((size_t)N_NODES * HEADS_C * 4);
  float*     denom = (float*)    carve((size_t)N_NODES * HEADS_C * 4);
  float*     agg   = (float*)    carve((size_t)N_NODES * HEADS_C * DIM_C * 4);
  _Float16*  gat16 = (_Float16*) carve((size_t)N_NODES * HEADS_C * DIM_C * 2);
  _Float16*  mid16 = (_Float16*) carve((size_t)N_NODES * DMLP_C * 2);
  _Float16*  Wl16  = (_Float16*) carve((size_t)DIM_C * HEADS_C * DIM_C * 2);
  _Float16*  Wr16  = (_Float16*) carve((size_t)DIM_C * HEADS_C * DIM_C * 2);
  _Float16*  Wp16  = (_Float16*) carve((size_t)HEADS_C * DIM_C * DIM_C * 2);
  _Float16*  W116  = (_Float16*) carve((size_t)DIM_C * DMLP_C * 2);
  _Float16*  W216  = (_Float16*) carve((size_t)DMLP_C * DIM_C * 2);

  hipMemcpyAsync(xbuf, x_in, (size_t)N_NODES * DIM_C * 4, hipMemcpyDeviceToDevice, stream);

  const int nWl = DIM_C * HEADS_C * DIM_C;   // 32768
  const int nWp = HEADS_C * DIM_C * DIM_C;   // 32768
  const int nW1 = DIM_C * DMLP_C;            // 65536
  const int nW2 = DMLP_C * DIM_C;            // 65536

  auto gemm_blocks = [](int M, int Nout) { return (unsigned)(((M >> 4) * (Nout >> 5) + 7) / 8); };

  for (int l = 0; l < DEPTH_C; ++l) {
    // weight packing into WMMA B-fragment order (f16)
    pack_w<<<(nWl + 255) / 256, 256, 0, stream>>>(Wl + (size_t)l * nWl, Wl16, DIM_C, HEADS_C * DIM_C);
    pack_w<<<(nWl + 255) / 256, 256, 0, stream>>>(Wr + (size_t)l * nWl, Wr16, DIM_C, HEADS_C * DIM_C);
    pack_w<<<(nWp + 255) / 256, 256, 0, stream>>>(Wp + (size_t)l * nWp, Wp16, HEADS_C * DIM_C, DIM_C);
    pack_w<<<(nW1 + 255) / 256, 256, 0, stream>>>(W1 + (size_t)l * nW1, W116, DIM_C, DMLP_C);
    pack_w<<<(nW2 + 255) / 256, 256, 0, stream>>>(W2 + (size_t)l * nW2, W216, DMLP_C, DIM_C);

    // --- GATv2 block ---
    layernorm_f16<<<(N_NODES + 7) / 8, 256, 0, stream>>>(
        xbuf, ln1_g + l * DIM_C, ln1_b + l * DIM_C, h16, N_NODES);
    gemm_wmma_f16<DIM_C, 0><<<gemm_blocks(N_NODES, HEADS_C * DIM_C), 256, 0, stream>>>(
        h16, Wl16, bl + (size_t)l * HEADS_C * DIM_C, xl, nullptr, N_NODES, HEADS_C * DIM_C);
    gemm_wmma_f16<DIM_C, 0><<<gemm_blocks(N_NODES, HEADS_C * DIM_C), 256, 0, stream>>>(
        h16, Wr16, br + (size_t)l * HEADS_C * DIM_C, xr, nullptr, N_NODES, HEADS_C * DIM_C);

    hipMemsetAsync(agg, 0, (size_t)N_NODES * HEADS_C * DIM_C * 4, stream);
    hipMemsetAsync(denom, 0, (size_t)N_NODES * HEADS_C * 4, stream);
    hipMemsetAsync(smax, 0, (size_t)N_NODES * HEADS_C * 4, stream);  // 0 == key(-NaN): identity

    edge_score<<<(E_EDGES + 7) / 8, 256, 0, stream>>>(
        srcI, dstI, xl, xr, att + (size_t)l * HEADS_C * DIM_C, sbuf, smax);
    edge_exp<<<(E_EDGES * HEADS_C + 255) / 256, 256, 0, stream>>>(dstI, sbuf, smax, denom);
    edge_agg<<<(E_EDGES + 7) / 8, 256, 0, stream>>>(srcI, dstI, xl, sbuf, denom, agg);

    gat_finalize<<<((N_NODES * HEADS_C * DIM_C) + 255) / 256, 256, 0, stream>>>(
        agg, gat_b + (size_t)l * HEADS_C * DIM_C, gat16, N_NODES * HEADS_C * DIM_C);
    // x += gat @ Wp + bp
    gemm_wmma_f16<HEADS_C * DIM_C, 1><<<gemm_blocks(N_NODES, DIM_C), 256, 0, stream>>>(
        gat16, Wp16, bp + (size_t)l * DIM_C, xbuf, nullptr, N_NODES, DIM_C);

    // --- MLP block ---
    layernorm_f16<<<(N_NODES + 7) / 8, 256, 0, stream>>>(
        xbuf, ln2_g + l * DIM_C, ln2_b + l * DIM_C, h16, N_NODES);
    gemm_wmma_f16<DIM_C, 2><<<gemm_blocks(N_NODES, DMLP_C), 256, 0, stream>>>(
        h16, W116, b1 + (size_t)l * DMLP_C, nullptr, mid16, N_NODES, DMLP_C);
    gemm_wmma_f16<DMLP_C, 1><<<gemm_blocks(N_NODES, DIM_C), 256, 0, stream>>>(
        mid16, W216, b2 + (size_t)l * DIM_C, xbuf, nullptr, N_NODES, DIM_C);
  }

  hipMemcpyAsync(d_out, xbuf, (size_t)N_NODES * DIM_C * 4, hipMemcpyDeviceToDevice, stream);
}
